// MSAPairWeightedAveragingFused_45749991637294
// MI455X (gfx1250) — compile-verified, matrix operand order checked by reading, not run on hardware
//
#include <hip/hip_runtime.h>

typedef __attribute__((ext_vector_type(16))) _Float16 v16h;
typedef __attribute__((ext_vector_type(8)))  _Float16 v8h;
typedef __attribute__((ext_vector_type(8)))  float    v8f;
typedef __attribute__((ext_vector_type(4)))  float    v4f;
typedef __attribute__((ext_vector_type(4)))  int      i32x4;
typedef __attribute__((ext_vector_type(8)))  int      i32x8;

#define S_DIM 512
#define R_DIM 384
#define CM 64      // C_MSA
#define CZ 128     // C_Z
#define CH 32      // C_HID
#define NH 8       // N_HEAD
#define HC 256     // NH*CH
#define INF_ 1e8f

// ---------- wave32 reductions ----------
__device__ __forceinline__ float wave_sum32(float v) {
#pragma unroll
  for (int o = 16; o > 0; o >>= 1) v += __shfl_xor(v, o, 32);
  return v;
}
__device__ __forceinline__ float wave_max32(float v) {
#pragma unroll
  for (int o = 16; o > 0; o >>= 1) v = fmaxf(v, __shfl_xor(v, o, 32));
  return v;
}

// Load a 16-half WMMA A/B fragment for one K-step from row-major data.
// Caller passes p = base + row*stride + kb, kb already includes (lane>=16)*8.
// ISA: lane holds K = {kb..kb+7} in elems 0..7 and {kb+16..kb+23} in 8..15.
__device__ __forceinline__ v16h load_frag16(const _Float16* p) {
  v8h lo = *(const v8h*)(p);
  v8h hi = *(const v8h*)(p + 16);
  v16h r;
#pragma unroll
  for (int e = 0; e < 8; ++e) { r[e] = lo[e]; r[e + 8] = hi[e]; }
  return r;
}

__device__ __forceinline__ v8f wmma_f16(v16h a, v16h b, v8f c) {
  return __builtin_amdgcn_wmma_f32_16x16x32_f16(false, a, false, b, (short)0, c,
                                                false, false);
}

// ---------------------------------------------------------------------------
// Kernel P: transpose weights to fp16, N-major (so B-fragments are contiguous)
// ---------------------------------------------------------------------------
__global__ void prep_weights(const float* __restrict__ wv,
                             const float* __restrict__ wg,
                             const float* __restrict__ wo,
                             _Float16* __restrict__ wvT,
                             _Float16* __restrict__ wgT,
                             _Float16* __restrict__ woT) {
  int idx = blockIdx.x * blockDim.x + threadIdx.x;   // 0..16383
  {  // wv/wg: [64 k][256 n] -> [256 n][64 k]
    int n = idx >> 6, k = idx & 63;
    wvT[idx] = (_Float16)wv[k * HC + n];
    wgT[idx] = (_Float16)wg[k * HC + n];
  }
  {  // wo: [256 k][64 n] -> [64 n][256 k]
    int n = idx >> 8, k = idx & 255;
    woT[idx] = (_Float16)wo[k * CM + n];
  }
}

// ---------------------------------------------------------------------------
// Kernel A: bias = LN(z) @ w_b + mask ; softmax over j ; W16[h][i][j] fp16
// ---------------------------------------------------------------------------
__global__ void bias_softmax_kernel(const float* __restrict__ z,
                                    const float* __restrict__ zmask,
                                    const float* __restrict__ lng,
                                    const float* __restrict__ lnb,
                                    const float* __restrict__ wb,
                                    _Float16* __restrict__ W16) {
  __shared__ float sb[R_DIM * NH];                   // 12 KB
  int i = blockIdx.x;
  int lane = threadIdx.x & 31, wid = threadIdx.x >> 5;

  float wbr[4][NH], lg[4], lb[4];
#pragma unroll
  for (int t = 0; t < 4; ++t) {
    lg[t] = lng[lane * 4 + t];
    lb[t] = lnb[lane * 4 + t];
#pragma unroll
    for (int h = 0; h < NH; ++h) wbr[t][h] = wb[(lane * 4 + t) * NH + h];
  }

  for (int j = wid; j < R_DIM; j += 8) {
    const float* zp = z + ((size_t)i * R_DIM + j) * CZ;
    v4f zv = *(const v4f*)(zp + lane * 4);
    float s  = zv.x + zv.y + zv.z + zv.w;
    float ss = zv.x * zv.x + zv.y * zv.y + zv.z * zv.z + zv.w * zv.w;
    s  = wave_sum32(s)  * (1.0f / CZ);
    ss = wave_sum32(ss) * (1.0f / CZ);
    float rstd = rsqrtf(ss - s * s + 1e-5f);
    float zn[4];
#pragma unroll
    for (int t = 0; t < 4; ++t) zn[t] = (zv[t] - s) * rstd * lg[t] + lb[t];
    float mb = INF_ * (zmask[(size_t)i * R_DIM + j] - 1.0f);
#pragma unroll
    for (int h = 0; h < NH; ++h) {
      float p = zn[0]*wbr[0][h] + zn[1]*wbr[1][h] + zn[2]*wbr[2][h] + zn[3]*wbr[3][h];
      p = wave_sum32(p);
      if (lane == 0) sb[j * NH + h] = p + mb;
    }
  }
  __syncthreads();

  int h = wid;                                       // wave per head
  float mx = -3.0e38f;
  for (int j = lane; j < R_DIM; j += 32) mx = fmaxf(mx, sb[j * NH + h]);
  mx = wave_max32(mx);
  float sum = 0.f;
  for (int j = lane; j < R_DIM; j += 32) sum += __expf(sb[j * NH + h] - mx);
  sum = wave_sum32(sum);
  float inv = __builtin_amdgcn_rcpf(sum);
  _Float16* Wp = W16 + ((size_t)h * R_DIM + i) * R_DIM;
  for (int j = lane; j < R_DIM; j += 32)
    Wp[j] = (_Float16)(__expf(sb[j * NH + h] - mx) * inv);
}

// ---------------------------------------------------------------------------
// Kernel B: m_ln = LN(m); v = mask*(m_ln@w_v) -> Vt[s][h][c][r] fp16
//           g = sigmoid(m_ln@w_g)             -> Gt[s][h][c][r] fp16
// Both outputs transposed so the einsum kernel gets contiguous fragments.
// ---------------------------------------------------------------------------
__global__ void vg_kernel(const float* __restrict__ m,
                          const float* __restrict__ mmask,
                          const float* __restrict__ lng,
                          const float* __restrict__ lnb,
                          const _Float16* __restrict__ wvT,
                          const _Float16* __restrict__ wgT,
                          _Float16* __restrict__ Vt,
                          _Float16* __restrict__ Gt) {
  __shared__ _Float16 sm[16 * 64];
  __shared__ float smask[16];
  int bt = blockIdx.x;
  int s = bt / 24, j0 = (bt % 24) * 16;
  int lane = threadIdx.x & 31, wid = threadIdx.x >> 5;
  size_t rowbase = (size_t)s * R_DIM + j0;

  // layernorm 16 rows into LDS fp16 (wave handles 2 rows)
#pragma unroll
  for (int rr = 0; rr < 2; ++rr) {
    int r = wid * 2 + rr;
    const float* mp = m + (rowbase + r) * CM;
    float x0 = mp[lane * 2], x1 = mp[lane * 2 + 1];
    float mu = wave_sum32(x0 + x1) * (1.0f / CM);
    float s2 = wave_sum32(x0 * x0 + x1 * x1) * (1.0f / CM);
    float rstd = rsqrtf(s2 - mu * mu + 1e-5f);
    sm[r * 64 + lane * 2]     = (_Float16)((x0 - mu) * rstd * lng[lane * 2]     + lnb[lane * 2]);
    sm[r * 64 + lane * 2 + 1] = (_Float16)((x1 - mu) * rstd * lng[lane * 2 + 1] + lnb[lane * 2 + 1]);
  }
  if (threadIdx.x < 16) smask[threadIdx.x] = mmask[rowbase + threadIdx.x];
  __syncthreads();

  int row = lane & 15, ks = (lane >> 4) * 8, mlo = ks;
  v16h a0 = load_frag16(&sm[row * 64 + 0  + ks]);
  v16h a1 = load_frag16(&sm[row * 64 + 32 + ks]);

#pragma unroll
  for (int q = 0; q < 4; ++q) {
    int tt = wid * 4 + q;                 // 0..15 -> v tiles, 16..31 -> g tiles
    bool isv = tt < 16;
    int n0 = isv ? tt * 16 : (tt - 16) * 16;
    const _Float16* wt = isv ? wvT : wgT;
    int n = n0 + row;
    v16h b0 = load_frag16(wt + (size_t)n * 64 + 0  + ks);
    v16h b1 = load_frag16(wt + (size_t)n * 64 + 32 + ks);
    v8f acc = {};
    acc = wmma_f16(a0, b0, acc);
    acc = wmma_f16(a1, b1, acc);
    int h = n >> 5, c = n & 31;
    size_t toff = (((size_t)s * NH + h) * CH + c) * R_DIM + j0 + mlo;
    v8h pk;
    if (isv) {
#pragma unroll
      for (int p = 0; p < 8; ++p) pk[p] = (_Float16)(acc[p] * smask[mlo + p]);
      *(v8h*)(Vt + toff) = pk;            // 8 consecutive r -> 16B store
    } else {
#pragma unroll
      for (int p = 0; p < 8; ++p)         // fast sigmoid via v_rcp_f32
        pk[p] = (_Float16)__builtin_amdgcn_rcpf(1.0f + __expf(-acc[p]));
      *(v8h*)(Gt + toff) = pk;
    }
  }
}

// ---------------------------------------------------------------------------
// Kernel C (fully fused): per s:
//   TDM-stage Vt[s] (192 KB contiguous) into LDS once;
//   for each of 24 i-tiles: 8 waves (one per head) do the einsum
//   (12 K-steps x 2 N-tiles WMMA), gate with Gt, transpose through LDS,
//   project through w_o slice (K=32 WMMA x 4 N-tiles), and cross-head
//   reduce the 16x64 fp32 output tile in LDS -> coalesced v4f stores.
// LDS: 192 KB V slab + 8 KB o-staging + 32 KB partials = 232 KB (<320 KB WGP).
// ---------------------------------------------------------------------------
__global__ void pwa_kernel(const _Float16* __restrict__ W16,
                           const _Float16* __restrict__ Vt,
                           const _Float16* __restrict__ Gt,
                           const _Float16* __restrict__ woT,
                           float* __restrict__ out) {
  extern __shared__ _Float16 slab[];
  _Float16* vslab = slab;                       // [NH][CH][R_DIM] = 98304 halves
  _Float16* otile = slab + (size_t)NH*CH*R_DIM; // [NH][16][32]   = 4096 halves
  float*    part  = (float*)(otile + NH*16*CH); // [NH][16][64]   = 8192 floats

  int s = blockIdx.x;
  int lane = threadIdx.x & 31, h = threadIdx.x >> 5;
  int row = lane & 15, ks = (lane >> 4) * 8, mlo = ks;

  // ---- Tensor Data Mover: Vt[s] (contiguous 192 KB) -> LDS ----
  if (h == 0) {
    uint64_t gaddr = (uint64_t)(const void*)(Vt + (size_t)s * NH * CH * R_DIM);
    uint32_t laddr = (uint32_t)(uintptr_t)(void*)vslab;    // LDS byte offset
    const uint32_t elems = (NH * CH * R_DIM) / 2;          // dwords = 49152
    i32x4 g0;
    g0[0] = 1;                                             // count=1, user desc
    g0[1] = (int)laddr;                                    // lds_addr
    g0[2] = (int)(uint32_t)(gaddr & 0xffffffffu);          // global_addr[31:0]
    g0[3] = (int)(((uint32_t)(gaddr >> 32) & 0x01ffffffu)  // global_addr[56:32]
                  | (2u << 30));                           // type = 2 (image)
    i32x8 g1;
    g1[0] = (int)(2u << 16);                   // data_size=2 -> 4-byte elements
    g1[1] = (int)((elems & 0xffffu) << 16);    // tensor_dim0[15:0] (bits 63:48)
    g1[2] = (int)((elems >> 16) | (1u << 16)); // tensor_dim0[31:16] | dim1=1
    g1[3] = (int)((elems & 0xffffu) << 16);    // tile_dim0 (bits 127:112)
    g1[4] = 0;                                 // tile_dim1/2 = 0 (1-D)
    g1[5] = (int)elems;                        // tensor_dim0_stride
    g1[6] = 0;
    g1[7] = 0;
    asm volatile("tensor_load_to_lds %0, %1" :: "s"(g0), "s"(g1) : "memory");
    __builtin_amdgcn_s_wait_tensorcnt(0);
  }
  __syncthreads();

  const _Float16* V0 = vslab + ((size_t)(h * CH) + row) * R_DIM;  // c = row
  const _Float16* V1 = V0 + (size_t)16 * R_DIM;                   // c = row+16
  const _Float16* Wh = W16 + (size_t)h * R_DIM * R_DIM;
  const _Float16* G0 = Gt + (((size_t)s * NH + h) * CH + row) * R_DIM;
  const _Float16* G1 = G0 + (size_t)16 * R_DIM;
  _Float16* ot = otile + h * (16 * CH);

  // Projection B-fragments: w_o rows k in [h*32, h*32+32), invariant per wave.
  v16h bw[4];
#pragma unroll
  for (int nt = 0; nt < 4; ++nt)
    bw[nt] = load_frag16(woT + (size_t)(nt * 16 + row) * HC + h * CH + ks);

  for (int it = 0; it < 24; ++it) {
    int i0 = it * 16;
    const _Float16* Wp = Wh + (size_t)(i0 + row) * R_DIM;
    v8f acc0 = {}, acc1 = {};
#pragma unroll 4
    for (int jj = 0; jj < R_DIM; jj += 32) {
      v16h a  = load_frag16(Wp + jj + ks);   // A from L2-resident weights
      v16h b0 = load_frag16(V0 + jj + ks);   // B from LDS (ds_load_b128)
      v16h b1 = load_frag16(V1 + jj + ks);
      acc0 = wmma_f16(a, b0, acc0);
      acc1 = wmma_f16(a, b1, acc1);
    }

    // gate (vector fp16 loads, layout mirrors Vt) and stage transposed o
    v8h ga = *(const v8h*)(G0 + i0 + mlo);
    v8h gb = *(const v8h*)(G1 + i0 + mlo);
#pragma unroll
    for (int p = 0; p < 8; ++p) {
      int M = mlo + p;
      ot[M * CH + row]      = (_Float16)(acc0[p] * (float)ga[p]);
      ot[M * CH + 16 + row] = (_Float16)(acc1[p] * (float)gb[p]);
    }

    // fused projection: o_tile[16x32] @ w_o[h-slice 32 x 64] (K=32, 4 N-tiles)
    v16h ao = load_frag16(ot + row * CH + ks);
    float* ph = part + h * (16 * CM);
#pragma unroll
    for (int nt = 0; nt < 4; ++nt) {
      v8f z = {};
      v8f pp = wmma_f16(ao, bw[nt], z);
#pragma unroll
      for (int p = 0; p < 8; ++p)
        ph[(mlo + p) * CM + nt * 16 + row] = pp[p];
    }
    __syncthreads();

    // cross-head reduction of the 16x64 fp32 tile; coalesced v4f store
    {
      int e = threadIdx.x * 4;                 // 1024 elems / 256 thr
      int mm = e >> 6, nn = e & 63;
      v4f a4 = {};
#pragma unroll
      for (int hh = 0; hh < NH; ++hh) {
        const float* pp = part + hh * (16 * CM) + e;
        a4.x += pp[0]; a4.y += pp[1]; a4.z += pp[2]; a4.w += pp[3];
      }
      *(v4f*)(out + ((size_t)s * R_DIM + i0 + mm) * CM + nn) = a4;
    }
    __syncthreads();                           // protect partials for next tile
  }
}

// ---------------------------------------------------------------------------
extern "C" void kernel_launch(void* const* d_in, const int* in_sizes, int n_in,
                              void* d_out, int out_size, void* d_ws, size_t ws_size,
                              hipStream_t stream) {
  const float* m     = (const float*)d_in[0];
  const float* z     = (const float*)d_in[1];
  const float* mmask = (const float*)d_in[2];
  const float* zmask = (const float*)d_in[3];
  const float* lnmg  = (const float*)d_in[4];
  const float* lnmb  = (const float*)d_in[5];
  const float* wv    = (const float*)d_in[6];
  const float* wg    = (const float*)d_in[7];
  const float* lnzg  = (const float*)d_in[8];
  const float* lnzb  = (const float*)d_in[9];
  const float* wb    = (const float*)d_in[10];
  const float* wo    = (const float*)d_in[11];
  float* out = (float*)d_out;

  char* ws = (char*)d_ws;
  size_t off = 0;
  auto alloc = [&](size_t bytes) {
    char* p = ws + off;
    off += (bytes + 255) & ~(size_t)255;
    return p;
  };
  _Float16* W16 = (_Float16*)alloc((size_t)NH * R_DIM * R_DIM * 2);        // 2.4 MB
  _Float16* Vt  = (_Float16*)alloc((size_t)S_DIM * NH * CH * R_DIM * 2);   // 100 MB
  _Float16* Gt  = (_Float16*)alloc((size_t)S_DIM * NH * CH * R_DIM * 2);   // 100 MB
  _Float16* wvT = (_Float16*)alloc(256 * 64 * 2);
  _Float16* wgT = (_Float16*)alloc(256 * 64 * 2);
  _Float16* woT = (_Float16*)alloc(64 * 256 * 2);

  prep_weights<<<64, 256, 0, stream>>>(wv, wg, wo, wvT, wgT, woT);
  bias_softmax_kernel<<<R_DIM, 256, 0, stream>>>(z, zmask, lnzg, lnzb, wb, W16);
  vg_kernel<<<S_DIM * 24, 256, 0, stream>>>(m, mmask, lnmg, lnmb, wvT, wgT, Vt, Gt);

  size_t lds_bytes = (size_t)NH * CH * R_DIM * sizeof(_Float16)   // 192 KB V slab
                   + (size_t)NH * 16 * CH * sizeof(_Float16)      // 8 KB o-staging
                   + (size_t)NH * 16 * CM * sizeof(float);        // 32 KB partials
  pwa_kernel<<<S_DIM, 256, lds_bytes, stream>>>(W16, Vt, Gt, woT, out);
}
